// CausalSelfAttention_35665408426715
// MI455X (gfx1250) — matrix operand order repaired and checked by reading, not compiled, not run on hardware
//
#include <hip/hip_runtime.h>
#include <hip/hip_bf16.h>

// ---------------------------------------------------------------------------
// CDNA5 (gfx1250) causal self-attention: bf16 WMMA (16x16x32), f32 accum,
// async global->LDS staging (ASYNCcnt) for the attention K/V stream.
// ---------------------------------------------------------------------------

typedef __bf16 bf16x8  __attribute__((ext_vector_type(8)));
typedef __bf16 bf16x16 __attribute__((ext_vector_type(16)));
typedef float  f32x8   __attribute__((ext_vector_type(8)));

#define NB 4
#define NN 2048
#define ND 1024
#define NH 16
#define HD 64
#define NM (NB * NN)        // 8192 tokens
#define N3D (3 * ND)        // 3072

__device__ __forceinline__ f32x8 wmma_bf16(bf16x16 a, bf16x16 b, f32x8 c) {
    return __builtin_amdgcn_wmma_f32_16x16x32_bf16(
        false, a, false, b, (short)0, c, false, false);
}

__device__ __forceinline__ bf16x16 frag_pair(const __bf16* lo, const __bf16* hi) {
    bf16x16 f;
    *(bf16x8*)&f       = *(const bf16x8*)lo;
    *((bf16x8*)&f + 1) = *(const bf16x8*)hi;
    return f;
}

// Async 16B global->LDS copy per lane (ASYNCcnt-tracked).
__device__ __forceinline__ void async_b128(unsigned lds_addr, const void* gaddr) {
    asm volatile("global_load_async_to_lds_b128 %0, %1, off"
                 :: "v"(lds_addr), "v"(gaddr) : "memory");
}
__device__ __forceinline__ void wait_async(void) {
    asm volatile("s_wait_asynccnt 0x0" ::: "memory");
}
__device__ __forceinline__ void wait_async_keep2(void) {
    asm volatile("s_wait_asynccnt 0x2" ::: "memory");
}

// reductions across the 16 lanes of a half-wave (one C-matrix row)
__device__ __forceinline__ float rowmax16(float x) {
    x = fmaxf(x, __shfl_xor(x, 1, 32));
    x = fmaxf(x, __shfl_xor(x, 2, 32));
    x = fmaxf(x, __shfl_xor(x, 4, 32));
    x = fmaxf(x, __shfl_xor(x, 8, 32));
    return x;
}
__device__ __forceinline__ float rowsum16(float x) {
    x += __shfl_xor(x, 1, 32);
    x += __shfl_xor(x, 2, 32);
    x += __shfl_xor(x, 4, 32);
    x += __shfl_xor(x, 8, 32);
    return x;
}

// ---------------------------------------------------------------------------
// Prologue conversions
// ---------------------------------------------------------------------------
__global__ void k_cvt_bf16(const float* __restrict__ in, __bf16* __restrict__ out, int n) {
    int i = (blockIdx.x * blockDim.x + threadIdx.x) * 4;
    if (i < n) {
        float4 v = *(const float4*)(in + i);
        out[i + 0] = (__bf16)v.x;
        out[i + 1] = (__bf16)v.y;
        out[i + 2] = (__bf16)v.z;
        out[i + 3] = (__bf16)v.w;
    }
}

// in[K][N] f32 -> out[N][K] bf16 (K-contiguous B-operand)
__global__ void k_transpose_cvt(const float* __restrict__ in, __bf16* __restrict__ out,
                                int K, int N) {
    __shared__ float tile[32][33];
    int kb = blockIdx.y * 32, nb = blockIdx.x * 32;
    int tx = threadIdx.x, ty = threadIdx.y;  // 32 x 8
#pragma unroll
    for (int r = 0; r < 32; r += 8)
        tile[ty + r][tx] = in[(size_t)(kb + ty + r) * N + nb + tx];
    __syncthreads();
#pragma unroll
    for (int r = 0; r < 32; r += 8)
        out[(size_t)(nb + ty + r) * K + kb + tx] = (__bf16)tile[tx][ty + r];
}

// ---------------------------------------------------------------------------
// Stage 1: QKV GEMM, wave tile 32x64 (8 WMMA / k-step), scatter epilogue:
//   Q -> [bh][n][hd] (pre-scaled by hd^-0.5), K -> [bh][n][hd], V -> [bh][hd][n]
// ---------------------------------------------------------------------------
__global__ __launch_bounds__(256) void k_gemm_qkv(
    const __bf16* __restrict__ A,  // [8192][1024]
    const __bf16* __restrict__ Bt, // [3072][1024]  (= Wqkv^T)
    __bf16* __restrict__ Qb, __bf16* __restrict__ Kb, __bf16* __restrict__ Vt) {
    int lane = threadIdx.x & 31, wave = threadIdx.x >> 5;
    int laneLo = lane & 15, laneHi = lane >> 4;
    int waveM = wave >> 1, waveN = wave & 1;
    int mBase = blockIdx.y * 128 + waveM * 32;
    int nBase = blockIdx.x * 128 + waveN * 64;

    f32x8 acc[2][4] = {};
    const __bf16* ar[2];
    const __bf16* bc[4];
#pragma unroll
    for (int i = 0; i < 2; i++) ar[i] = A + (size_t)(mBase + i * 16 + laneLo) * ND;
#pragma unroll
    for (int i = 0; i < 4; i++) bc[i] = Bt + (size_t)(nBase + i * 16 + laneLo) * ND;
    int aoff = laneHi * 8, boff = laneHi * 8;

    for (int kb = 0; kb < ND; kb += 32) {
        bf16x16 af[2], bf[4];
#pragma unroll
        for (int i = 0; i < 2; i++)
            af[i] = frag_pair(ar[i] + kb + aoff, ar[i] + kb + 16 + aoff);
#pragma unroll
        for (int i = 0; i < 4; i++)
            bf[i] = frag_pair(bc[i] + kb + boff, bc[i] + kb + 16 + boff);
#pragma unroll
        for (int sm = 0; sm < 2; sm++)
#pragma unroll
            for (int sn = 0; sn < 4; sn++)
                acc[sm][sn] = wmma_bf16(af[sm], bf[sn], acc[sm][sn]);
    }

#pragma unroll
    for (int sm = 0; sm < 2; sm++)
#pragma unroll
        for (int sn = 0; sn < 4; sn++) {
            int n = nBase + sn * 16 + laneLo;
            int t = n >> 10, hh = (n >> 6) & 15, d = n & 63;
#pragma unroll
            for (int r = 0; r < 8; r++) {
                int m = mBase + sm * 16 + r + laneHi * 8;
                int b = m >> 11, i = m & (NN - 1);
                float v = acc[sm][sn][r];
                size_t bh = (size_t)(b * NH + hh);
                if (t == 0)
                    Qb[(bh * NN + i) * HD + d] = (__bf16)(v * 0.125f);
                else if (t == 1)
                    Kb[(bh * NN + i) * HD + d] = (__bf16)v;
                else
                    Vt[(bh * HD + d) * NN + i] = (__bf16)v;
            }
        }
}

// ---------------------------------------------------------------------------
// Stage 2: causal flash attention. Block = 128 query rows of one (b,h);
// the 8 waves share double-buffered K/V tiles staged with async copies.
// ---------------------------------------------------------------------------
__global__ __launch_bounds__(256) void k_attn(
    const __bf16* __restrict__ Qb, const __bf16* __restrict__ Kb,
    const __bf16* __restrict__ Vt, __bf16* __restrict__ Ob) {
    __shared__ __align__(16) __bf16 Kl[2][32][64];  // [buf][key][d]   2 x 4KB
    __shared__ __align__(16) __bf16 Vl[2][64][32];  // [buf][d][key]   2 x 4KB
    __shared__ __align__(16) __bf16 Pl[8][16][32];  // per-wave P staging, 8KB

    int tid = threadIdx.x;
    int lane = tid & 31, wave = tid >> 5;
    int laneLo = lane & 15, laneHi = lane >> 4;
    int bh = blockIdx.y;
    int qb0 = blockIdx.x * 128;
    int qBase = qb0 + wave * 16;

    const __bf16* Qh = Qb + (size_t)bh * NN * HD;
    const __bf16* Kh = Kb + (size_t)bh * NN * HD;
    const __bf16* Vh = Vt + (size_t)bh * HD * NN;
    __bf16* Pw = &Pl[wave][0][0];

    unsigned ldsK[2] = {(unsigned)(uintptr_t)&Kl[0][0][0],
                        (unsigned)(uintptr_t)&Kl[1][0][0]};
    unsigned ldsV[2] = {(unsigned)(uintptr_t)&Vl[0][0][0],
                        (unsigned)(uintptr_t)&Vl[1][0][0]};
    // V staging addressing: thread copies 16B; 4 threads per d-row
    int vrow = tid >> 2, vseg = (tid & 3) * 8;  // vseg in elements

    // Q A-fragments for the full hd=64 (two 16x32 k-steps)
    const __bf16* qrow = Qh + (size_t)(qBase + laneLo) * HD;
    bf16x16 qa0 = frag_pair(qrow + laneHi * 8, qrow + 16 + laneHi * 8);
    bf16x16 qa1 = frag_pair(qrow + 32 + laneHi * 8, qrow + 48 + laneHi * 8);

    f32x8 acc[4] = {};
    float mrow[8], lrow[8];
#pragma unroll
    for (int r = 0; r < 8; r++) { mrow[r] = -3.0e38f; lrow[r] = 0.0f; }

    int nblkTotal = qb0 / 32 + 4;   // key blocks needed by the whole CTA
    int myLimit = qBase / 32 + 1;   // key blocks this wave actually uses

    // prologue: stage block 0 into buffer 0
    async_b128(ldsK[0] + tid * 16, (const char*)Kh + tid * 16);
    async_b128(ldsV[0] + vrow * 64 + vseg * 2,
               (const char*)(Vh + (size_t)vrow * NN + vseg));

    for (int blk = 0; blk < nblkTotal; blk++) {
        int buf = blk & 1;
        if (blk + 1 < nblkTotal) {  // stage next block into the other buffer
            int jn = (blk + 1) * 32;
            async_b128(ldsK[buf ^ 1] + tid * 16,
                       (const char*)(Kh + (size_t)jn * HD) + tid * 16);
            async_b128(ldsV[buf ^ 1] + vrow * 64 + vseg * 2,
                       (const char*)(Vh + (size_t)vrow * NN + jn + vseg));
            wait_async_keep2();  // drain previous stage; leave the 2 new in flight
        } else {
            wait_async();
        }
        __syncthreads();

        if (blk < myLimit) {   // wave-uniform; EXEC stays all-ones inside
            int jb = blk * 32;
            f32x8 s[2];
#pragma unroll
            for (int half = 0; half < 2; half++) {
                const __bf16* kcol = &Kl[buf][half * 16 + laneLo][0];
                bf16x16 b0 = frag_pair(kcol + laneHi * 8, kcol + 16 + laneHi * 8);
                bf16x16 b1 = frag_pair(kcol + 32 + laneHi * 8, kcol + 48 + laneHi * 8);
                f32x8 z = {};
                f32x8 t = wmma_bf16(qa0, b0, z);
                s[half] = wmma_bf16(qa1, b1, t);
            }
            if (jb + 32 > qBase) {  // diagonal block: causal mask
#pragma unroll
                for (int half = 0; half < 2; half++) {
                    int j = jb + half * 16 + laneLo;
#pragma unroll
                    for (int r = 0; r < 8; r++)
                        if (j > qBase + r + laneHi * 8) s[half][r] = -3.0e38f;
                }
            }
            float alpha[8];
#pragma unroll
            for (int r = 0; r < 8; r++) {
                float mt = rowmax16(fmaxf(s[0][r], s[1][r]));
                float mnew = fmaxf(mrow[r], mt);
                alpha[r] = __expf(mrow[r] - mnew);
                mrow[r] = mnew;
            }
#pragma unroll
            for (int r = 0; r < 8; r++) {
                float p0 = __expf(s[0][r] - mrow[r]);
                float p1 = __expf(s[1][r] - mrow[r]);
                int m = r + laneHi * 8;
                Pw[m * 32 + laneLo] = (__bf16)p0;
                Pw[m * 32 + 16 + laneLo] = (__bf16)p1;
                lrow[r] = lrow[r] * alpha[r] + rowsum16(p0 + p1);
                acc[0][r] *= alpha[r];
                acc[1][r] *= alpha[r];
                acc[2][r] *= alpha[r];
                acc[3][r] *= alpha[r];
            }
            const __bf16* prow = Pw + laneLo * 32;
            bf16x16 pf = frag_pair(prow + laneHi * 8, prow + 16 + laneHi * 8);
#pragma unroll
            for (int t = 0; t < 4; t++) {
                const __bf16* vcol = &Vl[buf][t * 16 + laneLo][0];
                bf16x16 vb = frag_pair(vcol + laneHi * 8, vcol + 16 + laneHi * 8);
                acc[t] = wmma_bf16(pf, vb, acc[t]);
            }
        }
        __syncthreads();  // everyone done with buf before it is overwritten
    }

    int b = bh >> 4, h = bh & 15;
#pragma unroll
    for (int r = 0; r < 8; r++) {
        int q = qBase + r + laneHi * 8;
        float inv = 1.0f / lrow[r];
        size_t base = ((size_t)(b * NN + q)) * ND + h * HD + laneLo;
#pragma unroll
        for (int t = 0; t < 4; t++)
            Ob[base + t * 16] = (__bf16)(acc[t][r] * inv);
    }
}

// ---------------------------------------------------------------------------
// Stage 3: output projection, wave tile 32x64  (O @ proj_w + bias -> f32)
// ---------------------------------------------------------------------------
__global__ __launch_bounds__(256) void k_gemm_proj(
    const __bf16* __restrict__ A,   // [8192][1024]
    const __bf16* __restrict__ Bt,  // [1024][1024]  (= proj_w^T)
    const float* __restrict__ bias, float* __restrict__ out) {
    int lane = threadIdx.x & 31, wave = threadIdx.x >> 5;
    int laneLo = lane & 15, laneHi = lane >> 4;
    int waveM = wave >> 1, waveN = wave & 1;
    int mBase = blockIdx.y * 128 + waveM * 32;
    int nBase = blockIdx.x * 128 + waveN * 64;

    f32x8 acc[2][4] = {};
    const __bf16* ar[2];
    const __bf16* bc[4];
#pragma unroll
    for (int i = 0; i < 2; i++) ar[i] = A + (size_t)(mBase + i * 16 + laneLo) * ND;
#pragma unroll
    for (int i = 0; i < 4; i++) bc[i] = Bt + (size_t)(nBase + i * 16 + laneLo) * ND;
    int aoff = laneHi * 8, boff = laneHi * 8;

    for (int kb = 0; kb < ND; kb += 32) {
        bf16x16 af[2], bf[4];
#pragma unroll
        for (int i = 0; i < 2; i++)
            af[i] = frag_pair(ar[i] + kb + aoff, ar[i] + kb + 16 + aoff);
#pragma unroll
        for (int i = 0; i < 4; i++)
            bf[i] = frag_pair(bc[i] + kb + boff, bc[i] + kb + 16 + boff);
#pragma unroll
        for (int sm = 0; sm < 2; sm++)
#pragma unroll
            for (int sn = 0; sn < 4; sn++)
                acc[sm][sn] = wmma_bf16(af[sm], bf[sn], acc[sm][sn]);
    }

#pragma unroll
    for (int sm = 0; sm < 2; sm++)
#pragma unroll
        for (int sn = 0; sn < 4; sn++) {
            int n = nBase + sn * 16 + laneLo;
            float bv = bias[n];
#pragma unroll
            for (int r = 0; r < 8; r++) {
                int m = mBase + sm * 16 + r + laneHi * 8;
                out[(size_t)m * ND + n] = acc[sm][sn][r] + bv;
            }
        }
}

// ---------------------------------------------------------------------------
// Launch
// ---------------------------------------------------------------------------
extern "C" void kernel_launch(void* const* d_in, const int* in_sizes, int n_in,
                              void* d_out, int out_size, void* d_ws, size_t ws_size,
                              hipStream_t stream) {
    const float* x      = (const float*)d_in[0];
    const float* qkv_w  = (const float*)d_in[1];
    const float* proj_w = (const float*)d_in[2];
    const float* proj_b = (const float*)d_in[3];
    float* out = (float*)d_out;

    char* ws = (char*)d_ws;
    __bf16* Xb  = (__bf16*)(ws + 0);          // 16 MiB
    __bf16* WQT = (__bf16*)(ws + 16777216);   //  6 MiB
    __bf16* PWT = (__bf16*)(ws + 23068672);   //  2 MiB
    __bf16* Qb  = (__bf16*)(ws + 25165824);   // 16 MiB
    __bf16* Kb  = (__bf16*)(ws + 41943040);   // 16 MiB
    __bf16* Vt  = (__bf16*)(ws + 58720256);   // 16 MiB
    __bf16* Ob  = (__bf16*)(ws + 75497472);   // 16 MiB   (total 88 MiB)

    dim3 tb(32, 8);
    k_cvt_bf16<<<(NM * ND) / (256 * 4), 256, 0, stream>>>(x, Xb, NM * ND);
    k_transpose_cvt<<<dim3(N3D / 32, ND / 32), tb, 0, stream>>>(qkv_w, WQT, ND, N3D);
    k_transpose_cvt<<<dim3(ND / 32, ND / 32), tb, 0, stream>>>(proj_w, PWT, ND, ND);

    k_gemm_qkv<<<dim3(N3D / 128, NM / 128), 256, 0, stream>>>(Xb, WQT, Qb, Kb, Vt);
    k_attn<<<dim3(NN / 128, NB * NH), 256, 0, stream>>>(Qb, Kb, Vt, Ob);
    k_gemm_proj<<<dim3(ND / 128, NM / 128), 256, 0, stream>>>(Ob, PWT, proj_b, out);
}
